// DEHNN_28226525069578
// MI455X (gfx1250) — matrix operand order repaired and checked by reading, not compiled, not run on hardware
//
#include <hip/hip_runtime.h>
#include <hip/hip_bf16.h>

typedef __attribute__((ext_vector_type(16))) _Float16 v16h;
typedef __attribute__((ext_vector_type(8)))  _Float16 v8h;
typedef __attribute__((ext_vector_type(8)))  float    v8f;

static inline long cdivl(long a, long b){ return (a + b - 1) / b; }

// ---------------- utility kernels ----------------

__global__ void k_fill_f32(float* __restrict__ p, float v, long n){
  long i = (long)blockIdx.x * blockDim.x + threadIdx.x;
  if (i < n) p[i] = v;
}

__global__ void k_f32_to_f16(const float* __restrict__ a, _Float16* __restrict__ o, long n){
  long i = (long)blockIdx.x * blockDim.x + threadIdx.x;
  if (i < n) o[i] = (_Float16)a[i];
}

__global__ void k_deg(const int* __restrict__ sn, const int* __restrict__ se,
                      float* __restrict__ dn, float* __restrict__ de, int ek){
  int e = blockIdx.x * blockDim.x + threadIdx.x;
  if (e < ek){ atomicAdd(dn + sn[e], 1.0f); atomicAdd(de + se[e], 1.0f); }
}

__global__ void k_rsqrt_max1(float* __restrict__ p, long n){
  long i = (long)blockIdx.x * blockDim.x + threadIdx.x;
  if (i < n) p[i] = rsqrtf(fmaxf(p[i], 1.0f));
}

__global__ void k_ew(const int* __restrict__ sn, const int* __restrict__ se,
                     const float* __restrict__ wn, const float* __restrict__ we,
                     float* __restrict__ ew, int ek){
  int e = blockIdx.x * blockDim.x + threadIdx.x;
  if (e < ek) ew[e] = wn[sn[e]] * we[se[e]];
}

__global__ void k_cnt(const int* __restrict__ batch, float* __restrict__ cnt, int n){
  int i = blockIdx.x * blockDim.x + threadIdx.x;
  if (i < n) atomicAdd(cnt + batch[i], 1.0f);
}

__global__ void k_vn_init(float* __restrict__ vn, const float* __restrict__ vn_emb, int total){
  int i = blockIdx.x * blockDim.x + threadIdx.x;
  if (i < total) vn[i] = vn_emb[i & 63];
}

// x[n,d] = node_l[n,d](f16) + vn[batch[n],d]
__global__ void k_make_x(const _Float16* __restrict__ nodeL, const float* __restrict__ vn,
                         const int* __restrict__ batch, float* __restrict__ x, long total){
  long i = (long)blockIdx.x * blockDim.x + threadIdx.x;
  if (i < total){
    long node = i >> 6; int d = (int)(i & 63);
    x[i] = (float)nodeL[i] + vn[((long)batch[node] << 6) + d];
  }
}

// segment-sum x by batch into pool[VN,64]; LDS accumulation to dodge hot-slot atomics
__global__ void k_pool(const float* __restrict__ x, const int* __restrict__ batch,
                       float* __restrict__ pool, long n, long nodes_per_block){
  __shared__ float s[64 * 64];
  for (int i = threadIdx.x; i < 64*64; i += blockDim.x) s[i] = 0.f;
  __syncthreads();
  long nb = (long)blockIdx.x * nodes_per_block;
  long ne = nb + nodes_per_block; if (ne > n) ne = n;
  for (long idx = nb*64 + threadIdx.x; idx < ne*64; idx += blockDim.x){
    long node = idx >> 6; int d = (int)(idx & 63);
    atomicAdd(&s[(batch[node] << 6) + d], x[idx]);
  }
  __syncthreads();
  for (int i = threadIdx.x; i < 64*64; i += blockDim.x){
    float v = s[i];
    if (v != 0.f) atomicAdd(&pool[i], v);
  }
}

__global__ void k_init_agg(const _Float16* __restrict__ net, float* __restrict__ agg, long total){
  long i = (long)blockIdx.x * blockDim.x + threadIdx.x;
  if (i < total) agg[i] = (float)net[i];
}

// agg[em[e],:] += x[en[e],:]
__global__ void k_scatter_src(const int* __restrict__ en, const int* __restrict__ em,
                              const float* __restrict__ x, float* __restrict__ agg, long ne){
  long t = (long)blockIdx.x * blockDim.x + threadIdx.x;
  if (t < ne*16){
    long e = t >> 4; int d4 = (int)(t & 15) << 2;
    const float* xs = x + ((long)en[e] << 6) + d4;
    float4 v = *(const float4*)xs;
    float* ag = agg + ((long)em[e] << 6) + d4;
    atomicAdd(ag+0, v.x); atomicAdd(ag+1, v.y); atomicAdd(ag+2, v.z); atomicAdd(ag+3, v.w);
  }
}

// agg[em[e],:] += ew[e] * x[en[e],:]
__global__ void k_scatter_sink(const int* __restrict__ en, const int* __restrict__ em,
                               const float* __restrict__ x, const float* __restrict__ ew,
                               float* __restrict__ agg, long ne){
  long t = (long)blockIdx.x * blockDim.x + threadIdx.x;
  if (t < ne*16){
    long e = t >> 4; int d4 = (int)(t & 15) << 2;
    float w = ew[e];
    const float* xs = x + ((long)en[e] << 6) + d4;
    float4 v = *(const float4*)xs;
    float* ag = agg + ((long)em[e] << 6) + d4;
    atomicAdd(ag+0, v.x*w); atomicAdd(ag+1, v.y*w); atomicAdd(ag+2, v.z*w); atomicAdd(ag+3, v.w*w);
  }
}

// back[sn[e],:] += ew[e] * h_net[se[e],:]
__global__ void k_scatter_back(const int* __restrict__ sn, const int* __restrict__ se,
                               const _Float16* __restrict__ hnet, const float* __restrict__ ew,
                               float* __restrict__ back, long ne){
  long t = (long)blockIdx.x * blockDim.x + threadIdx.x;
  if (t < ne*16){
    long e = t >> 4; int d4 = (int)(t & 15) << 2;
    float w = ew[e];
    const _Float16* hs = hnet + ((long)se[e] << 6) + d4;
    float* bk = back + ((long)sn[e] << 6) + d4;
    atomicAdd(bk+0, (float)hs[0]*w); atomicAdd(bk+1, (float)hs[1]*w);
    atomicAdd(bk+2, (float)hs[2]*w); atomicAdd(bk+3, (float)hs[3]*w);
  }
}

__global__ void k_add_f16(const float* __restrict__ a, const float* __restrict__ b,
                          _Float16* __restrict__ o, long n){
  long i = (long)blockIdx.x * blockDim.x + threadIdx.x;
  if (i < n) o[i] = (_Float16)(a[i] + b[i]);
}

// vn_out = vn + relu((pool/cnt + vn) @ W + b); tiny (64x64 @ 64x64), one block
__global__ void k_vn_update(const float* __restrict__ vn, const float* __restrict__ pool,
                            const float* __restrict__ cnt, const float* __restrict__ W,
                            const float* __restrict__ b, float* __restrict__ vn_out){
  __shared__ float temp[64 * 64];
  for (int i = threadIdx.x; i < 64*64; i += blockDim.x){
    int v = i >> 6;
    temp[i] = pool[i] / fmaxf(cnt[v], 1.0f) + vn[i];
  }
  __syncthreads();
  for (int i = threadIdx.x; i < 64*64; i += blockDim.x){
    int v = i >> 6, d = i & 63;
    float acc = b[d];
    for (int k = 0; k < 64; k++) acc += temp[(v << 6) + k] * W[k*64 + d];
    vn_out[i] = vn[i] + fmaxf(acc, 0.0f);
  }
}

// ret[n] = cat[n,:] @ out_w + out_b   (1 output column -> plain dot, weights in LDS)
__global__ void k_ret(const _Float16* __restrict__ nodeL, long blkstride,
                      const float* __restrict__ ow, const float* __restrict__ ob,
                      float* __restrict__ out, long n){
  __shared__ float w[256];
  for (int i = threadIdx.x; i < 256; i += blockDim.x) w[i] = ow[i];
  __syncthreads();
  long i = (long)blockIdx.x * blockDim.x + threadIdx.x;
  if (i < n){
    float acc = ob[0];
    for (int c = 0; c < 4; c++){
      const _Float16* p = nodeL + c*blkstride + (i << 6);
      for (int k = 0; k < 64; k++) acc += (float)p[k] * w[c*64 + k];
    }
    out[i] = acc;
  }
}

// ---------------- WMMA GEMM ----------------
// out[rows, *] = act( A[rows, K] @ B[K, ncolsB] + bias ), A element (r,k) at
// A[r*a_rs + (k>>6)*a_blk + (k&63)] (supports blocked "cat" layouts).
// Block: 256 thr = 8 waves; 128 rows x 64 cols; wave w = rows [16w,16w+16),
// 4 col-tiles of 16 via v_wmma_f32_16x16x32_f16.
// Fast path: f16 A, full tile in-bounds, K%32==0 -> A staged with
// global_load_async_to_lds_b128 (ASYNCcnt), no VGPR round-trip.
#define LDS_AS 48  // f16 row stride (96 B: 16B-aligned fragments, non-pow2 banks)

__global__ void k_gemm_wmma(const void* __restrict__ A, int a_is_f32, long a_blk, int a_rs,
                            const float* __restrict__ B, const float* __restrict__ bias,
                            long rows, int K, int ncolsB,
                            void* __restrict__ out, int out_is_f32, int out_stride,
                            int relu, int ncols_write){
  __shared__ __align__(16) _Float16 aLds[128 * LDS_AS];
  __shared__ __align__(16) _Float16 bLds[64 * LDS_AS];   // stored transposed: [col][k]

  const int tid  = threadIdx.x;
  const int wave = tid >> 5;
  const int lane = tid & 31;
  const int hh   = lane >> 4;   // lane half (0/1)
  const int nidx = lane & 15;   // output column within 16-tile / A row within tile
  const long r0  = (long)blockIdx.x * 128;
  const int  c0  = blockIdx.y * 64;

  // uniform per block: async tile copy legal (no padding/conversion needed)
  const bool fastA = (!a_is_f32) && ((K & 31) == 0) && (r0 + 128 <= rows);

  v8f acc[4];
  #pragma unroll
  for (int t = 0; t < 4; t++)
    #pragma unroll
    for (int j = 0; j < 8; j++) acc[t][j] = 0.0f;

  for (int kc = 0; kc < K; kc += 32){
    __syncthreads();
    // ---- stage A chunk: 128 rows x 32 k (f16) ----
    if (fastA){
      // 512 x 16B segments, 2 per thread, straight global -> LDS (ASYNCcnt)
      #pragma unroll
      for (int rep = 0; rep < 2; rep++){
        int idx  = tid + rep * 256;      // 0..511
        int arow = idx >> 2;             // 0..127
        int seg  = idx & 3;              // 4 x 16B per 64B row-chunk
        long elem = (r0 + arow) * (long)a_rs + (long)(kc >> 6) * a_blk + (kc & 63) + seg * 8;
        unsigned long long ga = (unsigned long long)(const _Float16*)A + (unsigned long long)elem * 2ull;
        unsigned lds = (unsigned)(__SIZE_TYPE__)(aLds + arow * LDS_AS + seg * 8);
        asm volatile("global_load_async_to_lds_b128 %0, %1, off"
                     :: "v"(lds), "v"(ga) : "memory");
      }
    } else {
      int arow = tid >> 1;            // 0..127
      int kh   = (tid & 1) * 16;      // 0 or 16
      long grow = r0 + arow;
      _Float16 tmp[16];
      if (grow < rows){
        #pragma unroll
        for (int e = 0; e < 16; e++){
          int k = kc + kh + e;
          float v = 0.0f;
          if (k < K){
            long addr = grow * (long)a_rs + (long)(k >> 6) * a_blk + (k & 63);
            v = a_is_f32 ? ((const float*)A)[addr] : (float)((const _Float16*)A)[addr];
          }
          tmp[e] = (_Float16)v;
        }
      } else {
        #pragma unroll
        for (int e = 0; e < 16; e++) tmp[e] = (_Float16)0.0f;
      }
      #pragma unroll
      for (int e = 0; e < 16; e++) aLds[arow * LDS_AS + kh + e] = tmp[e];
    }
    // ---- stage B chunk transposed: 64 cols x 32 k (f32 -> f16, zero-padded) ----
    for (int idx = tid; idx < 2048; idx += 256){
      int k = idx & 31, c = idx >> 5;
      int gk = kc + k, gc = c0 + c;
      float v = (gk < K && gc < ncolsB) ? B[(long)gk * ncolsB + gc] : 0.0f;
      bLds[c * LDS_AS + k] = (_Float16)v;
    }
    if (fastA) asm volatile("s_wait_asynccnt 0x0" ::: "memory");
    __syncthreads();

    // ---- A fragment: 16-bit A 16x32 layout: k = (e/8)*16 + hh*8 + e%8 ----
    v16h af;
    {
      const _Float16* ap = &aLds[(wave * 16 + nidx) * LDS_AS];
      v8h lo = *(const v8h*)(ap + hh * 8);
      v8h hi = *(const v8h*)(ap + 16 + hh * 8);
      #pragma unroll
      for (int e = 0; e < 8; e++){ af[e] = lo[e]; af[e + 8] = hi[e]; }
    }
    // ---- B fragments: 16-bit B 32x16 layout: k = (e/8)*16 + hh*8 + e%8, col = nidx ----
    #pragma unroll
    for (int ct = 0; ct < 4; ct++){
      const _Float16* bp = &bLds[(ct * 16 + nidx) * LDS_AS + hh * 8];
      v8h blo = *(const v8h*)bp;
      v8h bhi = *(const v8h*)(bp + 16);
      v16h bf;
      #pragma unroll
      for (int e = 0; e < 8; e++){ bf[e] = blo[e]; bf[e + 8] = bhi[e]; }
      acc[ct] = __builtin_amdgcn_wmma_f32_16x16x32_f16(
          /*neg_a=*/false, af, /*neg_b=*/false, bf,
          /*c_mod=*/(short)0, acc[ct], /*reuse_a=*/false, /*reuse_b=*/false);
    }
  }

  // ---- epilogue: C/D layout: lane holds col nidx, rows v + 8*hh ----
  #pragma unroll
  for (int ct = 0; ct < 4; ct++){
    int col = c0 + ct * 16 + nidx;
    if (col < ncols_write){
      float bv = bias ? bias[col] : 0.0f;
      #pragma unroll
      for (int v = 0; v < 8; v++){
        long grow = r0 + wave * 16 + v + 8 * hh;
        if (grow < rows){
          float val = acc[ct][v] + bv;
          if (relu) val = fmaxf(val, 0.0f);
          if (out_is_f32) ((float*)out)[grow * out_stride + col] = val;
          else            ((_Float16*)out)[grow * out_stride + col] = (_Float16)val;
        }
      }
    }
  }
}

static void launch_gemm(const void* A, int a_is_f32, long a_blk, int a_rs,
                        const float* B, const float* bias, long rows, int K, int ncolsB,
                        void* out, int out_is_f32, int out_stride, int relu, int ncols_write,
                        hipStream_t s){
  dim3 g((unsigned)cdivl(rows, 128), (unsigned)cdivl(ncolsB, 64));
  k_gemm_wmma<<<g, 256, 0, s>>>(A, a_is_f32, a_blk, a_rs, B, bias, rows, K, ncolsB,
                                out, out_is_f32, out_stride, relu, ncols_write);
}

// ---------------- host orchestration ----------------

extern "C" void kernel_launch(void* const* d_in, const int* in_sizes, int n_in,
                              void* d_out, int out_size, void* d_ws, size_t ws_size,
                              hipStream_t stream) {
  const float* node_features = (const float*)d_in[0];
  const float* net_features  = (const float*)d_in[1];
  const int*   sink_node     = (const int*)d_in[2];
  const int*   sink_net      = (const int*)d_in[3];
  const int*   src_node      = (const int*)d_in[4];
  const int*   src_net       = (const int*)d_in[5];
  const int*   batch         = (const int*)d_in[6];
  const float* w_node        = (const float*)d_in[8];
  const float* b_node        = (const float*)d_in[9];
  const float* w_net         = (const float*)d_in[10];
  const float* b_net         = (const float*)d_in[11];
  const float* vn_emb        = (const float*)d_in[12];
  const float* vn_mlp_w      = (const float*)d_in[13];
  const float* vn_mlp_b      = (const float*)d_in[14];
  const float* conv_net_w    = (const float*)d_in[15];
  const float* conv_net_b    = (const float*)d_in[16];
  const float* conv_node_w   = (const float*)d_in[17];
  const float* conv_node_b   = (const float*)d_in[18];
  const float* fc1_w         = (const float*)d_in[19];
  const float* fc1_b         = (const float*)d_in[20];
  const float* fc2_w         = (const float*)d_in[21];
  const float* fc2_b         = (const float*)d_in[22];
  const float* out_w         = (const float*)d_in[23];
  const float* out_b         = (const float*)d_in[24];

  const long N  = in_sizes[0] / 16;
  const long M  = in_sizes[1] / 16;
  const long EK = in_sizes[2];
  const long ES = in_sizes[4];
  const int  VN = 64;      // reference constant (num_vn lives on device)
  const int  NL = 3;       // L
  const long Nblk = N * 64;  // one 64-dim slab of cat, in elements

  // carve workspace
  char* ws = (char*)d_ws;
  size_t off = 0;
  auto carve = [&](size_t bytes) -> void* {
    void* p = ws + off;
    off += (bytes + 255) & ~(size_t)255;
    return p;
  };
  float*    w_n     = (float*)carve((size_t)N * 4);
  float*    w_e     = (float*)carve((size_t)M * 4);
  float*    ew      = (float*)carve((size_t)EK * 4);
  float*    cnt     = (float*)carve((size_t)VN * 4);
  float*    vn_a    = (float*)carve((size_t)VN * 64 * 4);
  float*    vn_b    = (float*)carve((size_t)VN * 64 * 4);
  float*    vn_pool = (float*)carve((size_t)VN * 64 * 4);
  float*    x       = (float*)carve((size_t)N * 64 * 4);
  float*    agg     = (float*)carve((size_t)M * 64 * 4);
  float*    back    = (float*)carve((size_t)N * 64 * 4);
  _Float16* nodeL   = (_Float16*)carve((size_t)4 * N * 64 * 2);
  _Float16* netA    = (_Float16*)carve((size_t)M * 64 * 2);
  _Float16* netB    = (_Float16*)carve((size_t)M * 64 * 2);
  _Float16* astage  = (_Float16*)carve((size_t)N * 64 * 2);
  _Float16* aggh    = (_Float16*)carve((size_t)M * 64 * 2);
  _Float16* fc1out  = (_Float16*)carve((size_t)N * 256 * 2);
  (void)ws_size; (void)n_in; (void)out_size;

  const int T = 256;
  #define GRID1(n) dim3((unsigned)cdivl((long)(n), T)), dim3(T)

  // ---- gcn_norm edge weights ----
  k_fill_f32<<<GRID1(N), 0, stream>>>(w_n, 0.f, N);
  k_fill_f32<<<GRID1(M), 0, stream>>>(w_e, 0.f, M);
  k_deg<<<GRID1(EK), 0, stream>>>(sink_node, sink_net, w_n, w_e, (int)EK);
  k_rsqrt_max1<<<GRID1(N), 0, stream>>>(w_n, N);
  k_rsqrt_max1<<<GRID1(M), 0, stream>>>(w_e, M);
  k_ew<<<GRID1(EK), 0, stream>>>(sink_node, sink_net, w_n, w_e, ew, (int)EK);

  // ---- batch counts + vn init ----
  k_fill_f32<<<GRID1(VN), 0, stream>>>(cnt, 0.f, VN);
  k_cnt<<<GRID1(N), 0, stream>>>(batch, cnt, (int)N);
  k_vn_init<<<GRID1(VN*64), 0, stream>>>(vn_a, vn_emb, VN*64);

  // ---- input embeddings (K=16 GEMMs, zero-padded to 32) ----
  launch_gemm(node_features, 1, 0, 16, w_node, b_node, N, 16, 64,
              nodeL, 0, 64, 1, 64, stream);
  launch_gemm(net_features, 1, 0, 16, w_net, b_net, M, 16, 64,
              netA, 0, 64, 1, 64, stream);

  float* vn_cur = vn_a; float* vn_nxt = vn_b;
  _Float16* net_cur = netA; _Float16* net_nxt = netB;

  for (int l = 0; l < NL; l++){
    // x = node_list[l] + vn[batch]
    k_make_x<<<GRID1(N*64), 0, stream>>>(nodeL + (long)l * Nblk, vn_cur, batch, x, N*64);
    if (l < NL - 1){
      k_fill_f32<<<GRID1(VN*64), 0, stream>>>(vn_pool, 0.f, VN*64);
      k_pool<<<dim3((unsigned)cdivl(N, 2048)), dim3(T), 0, stream>>>(x, batch, vn_pool, N, 2048);
    }
    // agg = net_list[l]; += src pins; += weighted sink pins
    k_init_agg<<<GRID1(M*64), 0, stream>>>(net_cur, agg, M*64);
    k_scatter_src<<<GRID1(ES*16), 0, stream>>>(src_node, src_net, x, agg, ES);
    k_scatter_sink<<<GRID1(EK*16), 0, stream>>>(sink_node, sink_net, x, ew, agg, EK);
    // h_net = relu(agg @ conv_net_w[l] + b); pre-convert agg to f16 (async-stageable A)
    k_f32_to_f16<<<GRID1(M*64), 0, stream>>>(agg, aggh, M*64);
    launch_gemm(aggh, 0, 0, 64, conv_net_w + (long)l*64*64, conv_net_b + (long)l*64,
                M, 64, 64, net_nxt, 0, 64, 1, 64, stream);
    // back = scatter(h_net[sink_net]*ew -> sink_node)
    k_fill_f32<<<GRID1(N*64), 0, stream>>>(back, 0.f, N*64);
    k_scatter_back<<<GRID1(EK*16), 0, stream>>>(sink_node, sink_net, net_nxt, ew, back, EK);
    // h_node = relu((x + back) @ conv_node_w[l] + b)
    k_add_f16<<<GRID1(N*64), 0, stream>>>(x, back, astage, N*64);
    launch_gemm(astage, 0, 0, 64, conv_node_w + (long)l*64*64, conv_node_b + (long)l*64,
                N, 64, 64, nodeL + (long)(l+1) * Nblk, 0, 64, 1, 64, stream);
    // vn update
    if (l < NL - 1){
      k_vn_update<<<dim3(1), dim3(T), 0, stream>>>(vn_cur, vn_pool, cnt,
          vn_mlp_w + (long)l*64*64, vn_mlp_b + (long)l*64, vn_nxt);
      float* t = vn_cur; vn_cur = vn_nxt; vn_nxt = t;
    }
    _Float16* t = net_cur; net_cur = net_nxt; net_nxt = t;
  }

  // ---- head: cat = [node0..node3] (blocked [4][N][64]) ----
  // fc1: [N,256] @ [256,256] -> relu f16
  launch_gemm(nodeL, 0, Nblk, 64, fc1_w, fc1_b, N, 256, 256,
              fc1out, 0, 256, 1, 256, stream);
  // fc2: [N,256] @ [256,8] -> relu f32 into d_out[0 : N*8]
  launch_gemm(fc1out, 0, 64, 256, fc2_w, fc2_b, N, 256, 8,
              d_out, 1, 8, 1, 8, stream);
  // ret: cat @ out_w + out_b into d_out[N*8 : N*9]
  k_ret<<<GRID1(N), 0, stream>>>(nodeL, Nblk, out_w, out_b, (float*)d_out + N*8, N);

  #undef GRID1
}
